// RSSM_13237089206882
// MI455X (gfx1250) — matrix operand order repaired and checked by reading, not compiled, not run on hardware
//
#include <hip/hip_runtime.h>
#include <math.h>

#define T_STEPS 64
#define BATCH   256
#define OBS_D   1024
#define ACT_D   32
#define S_DIM   256
#define H_DIM   512
#define GH_DIM  128
#define TB      (T_STEPS * BATCH)

typedef __attribute__((ext_vector_type(16))) unsigned short v16us;
typedef __attribute__((ext_vector_type(8)))  unsigned short v8us;
typedef __attribute__((ext_vector_type(16))) __bf16        v16bf;
typedef __attribute__((ext_vector_type(8)))  float         v8f;

// ---------------------------------------------------------------- helpers

__device__ __forceinline__ unsigned short f2bf(float f) {
  unsigned u = __builtin_bit_cast(unsigned, f);
  u += 0x7FFFu + ((u >> 16) & 1u);   // round-to-nearest-even
  return (unsigned short)(u >> 16);
}

__device__ __forceinline__ float sigmoidf_(float x) { return 1.0f / (1.0f + expf(-x)); }

// A fragment (16x32 bf16, contiguous k order, same as load_b32): lane l -> row m=l&15, half=l>>4;
// vec elems 0..15 = K = k0 + half*16 + 0..15
__device__ __forceinline__ v16bf load_a32(const unsigned short* A, int lda, int lane, int k0) {
  int m = lane & 15;
  int half = lane >> 4;
  const unsigned short* p = A + (size_t)m * lda + k0 + half * 16;
  v16us u = *(const v16us*)p;
  return __builtin_bit_cast(v16bf, u);
}

// B fragment (32x16 bf16) built from row-major weight W (N x K): B[k][n] = W[n][k].
// lane l -> column n = l&15; lanes 0-15 hold K=k0..k0+15, lanes 16-31 K=k0+16..k0+31.
__device__ __forceinline__ v16bf load_b32(const unsigned short* W, int ldw, int n0, int lane, int k0) {
  int n = lane & 15;
  int khalf = (lane >> 4) << 4;
  v16us u = *(const v16us*)(W + (size_t)(n0 + n) * ldw + k0 + khalf);
  return __builtin_bit_cast(v16bf, u);
}

// one 16(M) x 16(N) f32 tile of  A(16xK) * W(NxK)^T
__device__ __forceinline__ v8f gemm_tile16(const unsigned short* A, int lda,
                                           const unsigned short* W, int ldw,
                                           int n0, int K, int lane) {
  v8f c = {0.f, 0.f, 0.f, 0.f, 0.f, 0.f, 0.f, 0.f};
  for (int k0 = 0; k0 < K; k0 += 32) {
    v16bf a = load_a32(A, lda, lane, k0);
    v16bf b = load_b32(W, ldw, n0, lane, k0);
    c = __builtin_amdgcn_wmma_f32_16x16x32_bf16(false, a, false, b, (short)0, c, false, false);
  }
  return c;
}

// ---------------------------------------------------------------- small kernels

__global__ void conv_bf16_kernel(const float* __restrict__ in, unsigned short* __restrict__ out, int n) {
  int i = blockIdx.x * blockDim.x + threadIdx.x;
  if (i < n) out[i] = f2bf(in[i]);
}

// apre[tb][n] = act[tb] @ Wa^T + ba   (K=32 -> scalar f32, trivial cost)
__global__ void apre_kernel(const float* __restrict__ act, const float* __restrict__ Wa,
                            const float* __restrict__ ba, unsigned short* __restrict__ out) {
  int tb = blockIdx.x;
  int n = threadIdx.x;               // 256 threads
  const float* a = act + (size_t)tb * ACT_D;
  const float* w = Wa + (size_t)n * ACT_D;
  float s = ba[n];
#pragma unroll
  for (int k = 0; k < ACT_D; ++k) s += a[k] * w[k];
  out[(size_t)tb * 256 + n] = f2bf(s);
}

// generic tiled WMMA GEMM: out(MxN) = A(MxK,bf16) @ W(NxK,bf16)^T + bias
__global__ __launch_bounds__(512) void gemm_bf16_kernel(
    const unsigned short* __restrict__ A, const unsigned short* __restrict__ W,
    const float* __restrict__ bias, float* __restrict__ out, int M, int N, int K) {
  int wave = threadIdx.x >> 5, lane = threadIdx.x & 31;
  int ntiles = N >> 4;
  int idx = blockIdx.x * 16 + wave;
  int mt = idx / ntiles, nt = idx % ntiles;
  v8f c = gemm_tile16(A + (size_t)(mt * 16) * K, K, W, K, nt * 16, K, lane);
  int n = nt * 16 + (lane & 15);
  int mb = (lane >> 4) * 8;
#pragma unroll
  for (int r = 0; r < 8; ++r) out[(size_t)(mt * 16 + mb + r) * N + n] = c[r] + bias[n];
}

// ---------------------------------------------------------------- the sequential scan
// One workgroup (16 waves, 512 threads) owns 16 batch rows for all 64 steps.
// All activations stay in LDS; weights stream from L2 as bf16.

struct GBuf  { float gi[16][384]; float gh[16][384]; };
struct PEBuf { float p[16][512]; unsigned short e[16][512]; };
union  OvlU  { GBuf g; PEBuf pe; };

__global__ __launch_bounds__(512) void rssm_scan_kernel(
    const unsigned short* __restrict__ Ws_bf,    const float* __restrict__ bs,
    const unsigned short* __restrict__ Wih0_bf,  const float* __restrict__ bih0,
    const unsigned short* __restrict__ Whh0_bf,  const float* __restrict__ bhh0,
    const unsigned short* __restrict__ Wih1_bf,  const float* __restrict__ bih1,
    const unsigned short* __restrict__ Whh1_bf,  const float* __restrict__ bhh1,
    const unsigned short* __restrict__ Wph_bf,   const float* __restrict__ bph,
    const unsigned short* __restrict__ postW_bf, const float* __restrict__ post_bW,
    const float* __restrict__ post_g,            const float* __restrict__ post_b,
    const unsigned short* __restrict__ apre_bf,  const float* __restrict__ opre,
    const float* __restrict__ noise,
    const float* __restrict__ h_init,            const float* __restrict__ s_init,
    unsigned short* __restrict__ hs_bf,
    float* __restrict__ feat_out,                float* __restrict__ post_out)
{
  __shared__ __align__(32) unsigned short s_bf[16][256];   // s (bf16)
  __shared__ __align__(32) unsigned short x_bf[16][512];   // [s@Ws+bs , a@Wa+ba]
  __shared__ __align__(32) unsigned short h_bf[16][256];   // h (bf16), h0=[:128], h1=[128:]
  __shared__ __align__(32) float          h_f [16][256];   // h (f32)
  __shared__ __align__(32) OvlU           ovl;             // gates overlaid with p/e

  const int tid  = threadIdx.x;
  const int wave = tid >> 5;
  const int lane = tid & 31;
  const int rb   = blockIdx.x * 16;                        // batch-row block

  for (int i = tid; i < 16 * 256; i += 512) {
    int m = i >> 8, c = i & 255;
    float hv = h_init[(size_t)(rb + m) * 256 + c];
    h_f[m][c] = hv;
    h_bf[m][c] = f2bf(hv);
    s_bf[m][c] = f2bf(s_init[(size_t)(rb + m) * 256 + c]);
  }
  __syncthreads();

  for (int t = 0; t < T_STEPS; ++t) {
    const size_t rowbase = (size_t)t * BATCH + rb;

    // -- stage 1: x[:, :256] = s @ Ws^T + bs ; x[:, 256:] = apre[t]
    {
      int n0 = wave * 16;
      v8f c = gemm_tile16(&s_bf[0][0], 256, Ws_bf, 256, n0, 256, lane);
      int n = n0 + (lane & 15), mb = (lane >> 4) * 8;
#pragma unroll
      for (int r = 0; r < 8; ++r) x_bf[mb + r][n] = f2bf(c[r] + bs[n]);
    }
    for (int i = tid; i < 16 * 256; i += 512) {
      int m = i >> 8, c = i & 255;
      x_bf[m][256 + c] = apre_bf[(rowbase + m) * 256 + c];
    }
    __syncthreads();

    // -- stage 2: gi0 = x @ Wih0^T + bih0 (K=512) ; gh0 = h0 @ Whh0^T + bhh0 (K=128)
    for (int tt = 0; tt < 3; ++tt) {
      int tile = wave + 16 * tt;
      int nl = lane & 15, mb = (lane >> 4) * 8;
      if (tile < 24) {
        int n0 = tile * 16;
        v8f c = gemm_tile16(&x_bf[0][0], 512, Wih0_bf, 512, n0, 512, lane);
        int n = n0 + nl;
#pragma unroll
        for (int r = 0; r < 8; ++r) ovl.g.gi[mb + r][n] = c[r] + bih0[n];
      } else {
        int n0 = (tile - 24) * 16;
        v8f c = gemm_tile16(&h_bf[0][0], 256, Whh0_bf, 128, n0, 128, lane);
        int n = n0 + nl;
#pragma unroll
        for (int r = 0; r < 8; ++r) ovl.g.gh[mb + r][n] = c[r] + bhh0[n];
      }
    }
    __syncthreads();

    // -- GRU0 elementwise -> x1 into h[:, :128]
    for (int i = tid; i < 16 * 128; i += 512) {
      int m = i >> 7, j = i & 127;
      float r = sigmoidf_(ovl.g.gi[m][j] + ovl.g.gh[m][j]);
      float z = sigmoidf_(ovl.g.gi[m][128 + j] + ovl.g.gh[m][128 + j]);
      float n = tanhf(ovl.g.gi[m][256 + j] + r * ovl.g.gh[m][256 + j]);
      float x1 = (1.0f - z) * n + z * h_f[m][j];
      h_f[m][j] = x1;
      h_bf[m][j] = f2bf(x1);
    }
    __syncthreads();

    // -- stage 3: gi1 = x1 @ Wih1^T + bih1 ; gh1 = h1 @ Whh1^T + bhh1 (K=128 both)
    for (int tt = 0; tt < 3; ++tt) {
      int tile = wave + 16 * tt;
      int nl = lane & 15, mb = (lane >> 4) * 8;
      if (tile < 24) {
        int n0 = tile * 16;
        v8f c = gemm_tile16(&h_bf[0][0], 256, Wih1_bf, 128, n0, 128, lane);
        int n = n0 + nl;
#pragma unroll
        for (int r = 0; r < 8; ++r) ovl.g.gi[mb + r][n] = c[r] + bih1[n];
      } else {
        int n0 = (tile - 24) * 16;
        v8f c = gemm_tile16(&h_bf[0][128], 256, Whh1_bf, 128, n0, 128, lane);
        int n = n0 + nl;
#pragma unroll
        for (int r = 0; r < 8; ++r) ovl.g.gh[mb + r][n] = c[r] + bhh1[n];
      }
    }
    __syncthreads();

    // -- GRU1 elementwise -> x2 ; emit h_new (features cols 0:256 and hs_bf)
    for (int i = tid; i < 16 * 128; i += 512) {
      int m = i >> 7, j = i & 127;
      float r = sigmoidf_(ovl.g.gi[m][j] + ovl.g.gh[m][j]);
      float z = sigmoidf_(ovl.g.gi[m][128 + j] + ovl.g.gh[m][128 + j]);
      float n = tanhf(ovl.g.gi[m][256 + j] + r * ovl.g.gh[m][256 + j]);
      float x2 = (1.0f - z) * n + z * h_f[m][128 + j];
      h_f[m][128 + j] = x2;
      unsigned short x2b = f2bf(x2);
      h_bf[m][128 + j] = x2b;
      size_t row = rowbase + m;
      float x1v = h_f[m][j];
      feat_out[row * 512 + j] = x1v;
      feat_out[row * 512 + 128 + j] = x2;
      hs_bf[row * 256 + j] = h_bf[m][j];
      hs_bf[row * 256 + 128 + j] = x2b;
    }
    __syncthreads();

    // -- stage 4: p[:, :256] = h_new @ Wph^T + bph ; p[:, 256:] = opre[t]
    {
      int n0 = wave * 16;
      v8f c = gemm_tile16(&h_bf[0][0], 256, Wph_bf, 256, n0, 256, lane);
      int n = n0 + (lane & 15), mb = (lane >> 4) * 8;
#pragma unroll
      for (int r = 0; r < 8; ++r) ovl.pe.p[mb + r][n] = c[r] + bph[n];
    }
    for (int i = tid; i < 16 * 256; i += 512) {
      int m = i >> 8, c = i & 255;
      ovl.pe.p[m][256 + c] = opre[(rowbase + m) * 256 + c];
    }
    __syncthreads();

    // -- stage 5: per-row LayerNorm(512) + ELU (wave w -> row w)
    {
      int row = wave;
      float vals[16];
      float s1 = 0.f, s2 = 0.f;
#pragma unroll
      for (int i = 0; i < 16; ++i) {
        float v = ovl.pe.p[row][lane + 32 * i];
        vals[i] = v; s1 += v; s2 += v * v;
      }
#pragma unroll
      for (int off = 16; off >= 1; off >>= 1) {
        s1 += __shfl_xor(s1, off, 32);
        s2 += __shfl_xor(s2, off, 32);
      }
      float mean = s1 * (1.0f / 512.0f);
      float var  = s2 * (1.0f / 512.0f) - mean * mean;
      float rstd = rsqrtf(var + 1e-3f);
#pragma unroll
      for (int i = 0; i < 16; ++i) {
        int c = lane + 32 * i;
        float y = (vals[i] - mean) * rstd * post_g[c] + post_b[c];
        float e = y > 0.f ? y : expm1f(y);
        ovl.pe.e[row][c] = f2bf(e);
      }
    }
    __syncthreads();

    // -- stage 6: post = e @ postW^T + post_bW ; rsample -> s_new ; emit outputs
    {
      int n0 = wave * 16;                                   // wave owns mu tile n0 and sraw tile 256+n0
      v8f cmu = gemm_tile16(&ovl.pe.e[0][0], 512, postW_bf, 512, n0, 512, lane);
      v8f csr = gemm_tile16(&ovl.pe.e[0][0], 512, postW_bf, 512, 256 + n0, 512, lane);
      int n = n0 + (lane & 15), mb = (lane >> 4) * 8;
#pragma unroll
      for (int r = 0; r < 8; ++r) {
        int m = mb + r;
        size_t row = rowbase + m;
        float mu = cmu[r] + post_bW[n];
        float sr = csr[r] + post_bW[256 + n];
        post_out[row * 512 + n] = mu;
        post_out[row * 512 + 256 + n] = sr;
        float stdv = 0.1f + 0.1f * sigmoidf_(sr);           // MIN + (MAX-MIN)*sigmoid
        float sn = mu + stdv * noise[row * 256 + n];
        feat_out[row * 512 + 256 + n] = sn;
        s_bf[m][n] = f2bf(sn);
      }
    }
    __syncthreads();
  }
}

// ---------------------------------------------------------------- priors (parallel, fused)
// block g: rows r0..r0+15 of (T*B);  hs->GEMM(K=256) -> LN+ELU -> GEMM(K=512) -> priors

__global__ __launch_bounds__(512) void rssm_prior_kernel(
    const unsigned short* __restrict__ hs_bf,
    const unsigned short* __restrict__ Wprh_bf, const float* __restrict__ bprh,
    const float* __restrict__ prior_g,          const float* __restrict__ prior_b,
    const unsigned short* __restrict__ priorW_bf, const float* __restrict__ prior_bW,
    float* __restrict__ prior_out)
{
  __shared__ __align__(32) float          ph[16][512];
  __shared__ __align__(32) unsigned short eb[16][512];
  const int tid = threadIdx.x, wave = tid >> 5, lane = tid & 31;
  const size_t r0 = (size_t)blockIdx.x * 16;
  const unsigned short* A = hs_bf + r0 * 256;

  for (int tt = 0; tt < 2; ++tt) {
    int n0 = (wave + 16 * tt) * 16;
    v8f c = gemm_tile16(A, 256, Wprh_bf, 256, n0, 256, lane);
    int n = n0 + (lane & 15), mb = (lane >> 4) * 8;
#pragma unroll
    for (int r = 0; r < 8; ++r) ph[mb + r][n] = c[r] + bprh[n];
  }
  __syncthreads();

  {
    int row = wave;
    float vals[16], s1 = 0.f, s2 = 0.f;
#pragma unroll
    for (int i = 0; i < 16; ++i) {
      float v = ph[row][lane + 32 * i];
      vals[i] = v; s1 += v; s2 += v * v;
    }
#pragma unroll
    for (int off = 16; off >= 1; off >>= 1) {
      s1 += __shfl_xor(s1, off, 32);
      s2 += __shfl_xor(s2, off, 32);
    }
    float mean = s1 * (1.0f / 512.0f);
    float var  = s2 * (1.0f / 512.0f) - mean * mean;
    float rstd = rsqrtf(var + 1e-3f);
#pragma unroll
    for (int i = 0; i < 16; ++i) {
      int c = lane + 32 * i;
      float y = (vals[i] - mean) * rstd * prior_g[c] + prior_b[c];
      float e = y > 0.f ? y : expm1f(y);
      eb[row][c] = f2bf(e);
    }
  }
  __syncthreads();

  for (int tt = 0; tt < 2; ++tt) {
    int n0 = (wave + 16 * tt) * 16;
    v8f c = gemm_tile16(&eb[0][0], 512, priorW_bf, 512, n0, 512, lane);
    int n = n0 + (lane & 15), mb = (lane >> 4) * 8;
#pragma unroll
    for (int r = 0; r < 8; ++r)
      prior_out[(r0 + mb + r) * 512 + n] = c[r] + prior_bW[n];
  }
}

// ---------------------------------------------------------------- host orchestration

extern "C" void kernel_launch(void* const* d_in, const int* in_sizes, int n_in,
                              void* d_out, int out_size, void* d_ws, size_t ws_size,
                              hipStream_t stream) {
  (void)in_sizes; (void)n_in; (void)out_size; (void)ws_size;

  const float* obs     = (const float*)d_in[0];
  const float* act     = (const float*)d_in[1];
  const float* h_init  = (const float*)d_in[2];
  const float* s_init  = (const float*)d_in[3];
  const float* noise   = (const float*)d_in[4];
  const float* Ws      = (const float*)d_in[5];
  const float* bs      = (const float*)d_in[6];
  const float* Wa      = (const float*)d_in[7];
  const float* ba      = (const float*)d_in[8];
  const float* Wih0    = (const float*)d_in[9];
  const float* Whh0    = (const float*)d_in[10];
  const float* bih0    = (const float*)d_in[11];
  const float* bhh0    = (const float*)d_in[12];
  const float* Wih1    = (const float*)d_in[13];
  const float* Whh1    = (const float*)d_in[14];
  const float* bih1    = (const float*)d_in[15];
  const float* bhh1    = (const float*)d_in[16];
  const float* Wph     = (const float*)d_in[17];
  const float* bph     = (const float*)d_in[18];
  const float* Wpo     = (const float*)d_in[19];
  const float* bpo     = (const float*)d_in[20];
  const float* post_g  = (const float*)d_in[21];
  const float* post_b  = (const float*)d_in[22];
  const float* post_W  = (const float*)d_in[23];
  const float* post_bW = (const float*)d_in[24];
  const float* Wprh    = (const float*)d_in[25];
  const float* bprh    = (const float*)d_in[26];
  const float* prior_g = (const float*)d_in[27];
  const float* prior_b = (const float*)d_in[28];
  const float* prior_W = (const float*)d_in[29];
  const float* prior_bW= (const float*)d_in[30];

  float* feat_out  = (float*)d_out;                       // (T,B,512)
  float* prior_out = feat_out + (size_t)TB * 512;         // (T,B,512)
  float* post_out  = prior_out + (size_t)TB * 512;        // (T,B,512)

  char* wp = (char*)d_ws;
  auto carve = [&](size_t elems, size_t esize) -> void* {
    void* q = (void*)wp;
    wp += (elems * esize + 255) & ~(size_t)255;
    return q;
  };
  unsigned short* Ws_bf     = (unsigned short*)carve(256 * 256, 2);
  unsigned short* Wih0_bf   = (unsigned short*)carve(384 * 512, 2);
  unsigned short* Whh0_bf   = (unsigned short*)carve(384 * 128, 2);
  unsigned short* Wih1_bf   = (unsigned short*)carve(384 * 128, 2);
  unsigned short* Whh1_bf   = (unsigned short*)carve(384 * 128, 2);
  unsigned short* Wph_bf    = (unsigned short*)carve(256 * 256, 2);
  unsigned short* postW_bf  = (unsigned short*)carve(512 * 512, 2);
  unsigned short* Wprh_bf   = (unsigned short*)carve(512 * 256, 2);
  unsigned short* priorW_bf = (unsigned short*)carve(512 * 512, 2);
  unsigned short* Wpo_bf    = (unsigned short*)carve(256 * 1024, 2);
  unsigned short* obs_bf    = (unsigned short*)carve((size_t)TB * 1024, 2);
  unsigned short* apre_bf   = (unsigned short*)carve((size_t)TB * 256, 2);
  float*          opre      = (float*)carve((size_t)TB * 256, 4);
  unsigned short* hs_bf     = (unsigned short*)carve((size_t)TB * 256, 2);

  auto conv = [&](const float* src, unsigned short* dst, size_t n) {
    conv_bf16_kernel<<<(unsigned)((n + 255) / 256), 256, 0, stream>>>(src, dst, (int)n);
  };
  conv(Ws, Ws_bf, 256 * 256);
  conv(Wih0, Wih0_bf, 384 * 512);
  conv(Whh0, Whh0_bf, 384 * 128);
  conv(Wih1, Wih1_bf, 384 * 128);
  conv(Whh1, Whh1_bf, 384 * 128);
  conv(Wph, Wph_bf, 256 * 256);
  conv(post_W, postW_bf, 512 * 512);
  conv(Wprh, Wprh_bf, 512 * 256);
  conv(prior_W, priorW_bf, 512 * 512);
  conv(Wpo, Wpo_bf, 256 * 1024);
  conv(obs, obs_bf, (size_t)TB * 1024);

  // act projection (scan-independent half of x)
  apre_kernel<<<TB, 256, 0, stream>>>(act, Wa, ba, apre_bf);

  // obs projection (scan-independent half of p): 16384 x 1024 x 256 WMMA GEMM
  gemm_bf16_kernel<<<(TB / 16) * (256 / 16) / 16, 512, 0, stream>>>(
      obs_bf, Wpo_bf, bpo, opre, TB, 256, 1024);

  // sequential scan: 16 row-blocks, no cross-WG sync needed
  rssm_scan_kernel<<<16, 512, 0, stream>>>(
      Ws_bf, bs, Wih0_bf, bih0, Whh0_bf, bhh0, Wih1_bf, bih1, Whh1_bf, bhh1,
      Wph_bf, bph, postW_bf, post_bW, post_g, post_b,
      apre_bf, opre, noise, h_init, s_init, hs_bf, feat_out, post_out);

  // priors: fully parallel fused GEMM -> LN+ELU -> GEMM
  rssm_prior_kernel<<<TB / 16, 512, 0, stream>>>(
      hs_bf, Wprh_bf, bprh, prior_g, prior_b, priorW_bf, prior_bW, prior_out);
}
